// ScaleGraphBlock1_309237646113
// MI455X (gfx1250) — compile-verified, matrix operand order checked by reading, not run on hardware
//
#include <hip/hip_runtime.h>
#include <math.h>

#define BATCH   32
#define LSEQ    64
#define DMODEL  512
#define NODES   16
#define NODE_DIM 10
#define CONV_CH 32
#define SKIP_CH 32
#define ALPHA_MP 0.05f
#define D_STATE 16
#define D_INNER 1024
#define DT_RANK 32
#define KH      497
#define NTOK    (BATCH*LSEQ)   // 2048

// GEMM tiling: 64x128 block tile, 8 waves, each wave a 32x32 C sub-tile
#define BM  64
#define BN  128
#define BKK 32
#define LDH 40    // LDS row stride in halves (8-half pad; 80B rows stay 16B-aligned)

typedef __attribute__((ext_vector_type(16))) _Float16 v16h;
typedef __attribute__((ext_vector_type(8)))  _Float16 v8h;
typedef __attribute__((ext_vector_type(8)))  float    v8f;

// Generic batched GEMM: C[z] = A[z] (MxK) * B[z] (KxN) [+bias] [+act]
// transB: B element (k,n) read as Bp[n*ldb + k]  (i.e. C = A * W^T)
// biasMode: 0 none, 1 per-m, 2 per-n.  act: 0 none, 1 softplus, 2 exact gelu
// vecA/vecB: caller guarantees 16B-aligned rows (ld % 4 == 0, aligned base)
// Out-of-range rows/cols are CLAMPED (not masked): they only feed C entries the
// epilogue never stores. Only the K tail is zero-masked (cndmask, no EXEC churn).
// Double-buffered LDS: stage step s while issuing the 4 WMMAs of step s-1.
__global__ __launch_bounds__(256) void wmma_gemm(
        const float* __restrict__ A, const float* __restrict__ B,
        float* __restrict__ C, const float* __restrict__ bias,
        int M, int N, int K, int lda, int ldb, int ldc,
        int transB, int biasMode, int act, int vecA, int vecB,
        long long strideA, long long strideB, long long strideC)
{
    __shared__ __align__(16) _Float16 As[2][BM * LDH];   // 2 x 5120 B
    __shared__ __align__(16) _Float16 Bs[2][BN * LDH];   // 2 x 10240 B

    const int tid  = threadIdx.x;
    const int lane = tid & 31;
    const int wave = tid >> 5;
    const int r    = lane & 15;
    const int h    = lane >> 4;
    const int bn0  = blockIdx.x * BN;
    const int bm0  = blockIdx.y * BM;
    const long long z = blockIdx.z;
    const float* Ap = A + z * strideA;
    const float* Bp = B + z * strideB;
    float*       Cp = C + z * strideC;

    const int mt = wave >> 2;        // 0..1 -> 32-row band
    const int nt = wave & 3;         // 0..3 -> 32-col band

    v8f acc[4];
#pragma unroll
    for (int i = 0; i < 4; ++i) acc[i] = (v8f){0.f,0.f,0.f,0.f,0.f,0.f,0.f,0.f};

    // ---- staging plans (clamped rows/cols) ----
    const int a_m   = tid >> 2;                  // 0..63
    const int a_k   = (tid & 3) * 8;             // 0,8,16,24
    const long long a_row = (long long)min(bm0 + a_m, M - 1) * lda;

    const int bt_n0 = tid >> 2;                  // transB: n-rows t>>2 and +64
    const int bt_k  = (tid & 3) * 8;
    const long long bt_row0 = (long long)min(bn0 + bt_n0,      N - 1) * ldb;
    const long long bt_row1 = (long long)min(bn0 + bt_n0 + 64, N - 1) * ldb;

    const int bnt_k = tid >> 3;                  // no-trans: k-row 0..31
    const int bnt_n = (tid & 7) * 16;            // 16 n's per thread

    auto cvt8 = [](const float4& f0, const float4& f1) {
        v8h hv;
        hv[0]=(_Float16)f0.x; hv[1]=(_Float16)f0.y; hv[2]=(_Float16)f0.z; hv[3]=(_Float16)f0.w;
        hv[4]=(_Float16)f1.x; hv[5]=(_Float16)f1.y; hv[6]=(_Float16)f1.z; hv[7]=(_Float16)f1.w;
        return hv;
    };

    auto stage = [&](int kb, int buf) {
        if (kb + BKK <= K) {                     // ---- full step (uniform branch) ----
            // A tile
            {
                v8h hv;
                if (vecA) {
                    hv = cvt8(*(const float4*)(Ap + a_row + kb + a_k),
                              *(const float4*)(Ap + a_row + kb + a_k + 4));
                } else {
#pragma unroll
                    for (int j = 0; j < 8; ++j) hv[j] = (_Float16)Ap[a_row + kb + a_k + j];
                }
                *(v8h*)(&As[buf][a_m * LDH + a_k]) = hv;
            }
            // B tile -> Bt[n][k]
            if (transB) {
                v8h hv0, hv1;
                if (vecB) {
                    hv0 = cvt8(*(const float4*)(Bp + bt_row0 + kb + bt_k),
                               *(const float4*)(Bp + bt_row0 + kb + bt_k + 4));
                    hv1 = cvt8(*(const float4*)(Bp + bt_row1 + kb + bt_k),
                               *(const float4*)(Bp + bt_row1 + kb + bt_k + 4));
                } else {
#pragma unroll
                    for (int j = 0; j < 8; ++j) {
                        hv0[j] = (_Float16)Bp[bt_row0 + kb + bt_k + j];
                        hv1[j] = (_Float16)Bp[bt_row1 + kb + bt_k + j];
                    }
                }
                *(v8h*)(&Bs[buf][ bt_n0       * LDH + bt_k]) = hv0;
                *(v8h*)(&Bs[buf][(bt_n0 + 64) * LDH + bt_k]) = hv1;
            } else {
                const long long rowoff = (long long)(kb + bnt_k) * ldb;
                float f[16];
                if (vecB) {
#pragma unroll
                    for (int q = 0; q < 4; ++q) {
                        const float4 fv = *(const float4*)(Bp + rowoff + bn0 + bnt_n + q * 4);
                        f[q*4+0]=fv.x; f[q*4+1]=fv.y; f[q*4+2]=fv.z; f[q*4+3]=fv.w;
                    }
                } else {
#pragma unroll
                    for (int j = 0; j < 16; ++j) f[j] = Bp[rowoff + min(bn0 + bnt_n + j, N - 1)];
                }
#pragma unroll
                for (int j = 0; j < 16; ++j)
                    Bs[buf][(bnt_n + j) * LDH + bnt_k] = (_Float16)f[j];
            }
        } else {                                  // ---- K tail (zero-masked) ----
            {
                v8h hv;
#pragma unroll
                for (int j = 0; j < 8; ++j) {
                    const int k = kb + a_k + j;
                    const float v = Ap[a_row + ((k < K) ? k : (K - 1))];
                    hv[j] = (_Float16)((k < K) ? v : 0.f);
                }
                *(v8h*)(&As[buf][a_m * LDH + a_k]) = hv;
            }
            if (transB) {
                v8h hv0, hv1;
#pragma unroll
                for (int j = 0; j < 8; ++j) {
                    const int k = kb + bt_k + j;
                    const int kc = (k < K) ? k : (K - 1);
                    const float m0 = (k < K) ? 1.f : 0.f;
                    hv0[j] = (_Float16)(m0 * Bp[bt_row0 + kc]);
                    hv1[j] = (_Float16)(m0 * Bp[bt_row1 + kc]);
                }
                *(v8h*)(&Bs[buf][ bt_n0       * LDH + bt_k]) = hv0;
                *(v8h*)(&Bs[buf][(bt_n0 + 64) * LDH + bt_k]) = hv1;
            } else {
                const int gk = kb + bnt_k;
                const long long rowoff = (long long)((gk < K) ? gk : (K - 1)) * ldb;
                const float km = (gk < K) ? 1.f : 0.f;
#pragma unroll
                for (int j = 0; j < 16; ++j) {
                    const float v = Bp[rowoff + min(bn0 + bnt_n + j, N - 1)];
                    Bs[buf][(bnt_n + j) * LDH + bnt_k] = (_Float16)(km * v);
                }
            }
        }
    };

    auto ldfrag = [&](const _Float16* rowp) {
        const v8h lo = *(const v8h*)(rowp + h * 8);
        const v8h hi = *(const v8h*)(rowp + 16 + h * 8);
        return __builtin_shufflevector(lo, hi,
            0,1,2,3,4,5,6,7,8,9,10,11,12,13,14,15);
    };

    auto mma_step = [&](int buf) {
        v16h af[2], bf[2];
#pragma unroll
        for (int mi = 0; mi < 2; ++mi)
            af[mi] = ldfrag(&As[buf][(mt * 32 + mi * 16 + r) * LDH]);
#pragma unroll
        for (int ni = 0; ni < 2; ++ni)
            bf[ni] = ldfrag(&Bs[buf][(nt * 32 + ni * 16 + r) * LDH]);
        acc[0] = __builtin_amdgcn_wmma_f32_16x16x32_f16(false, af[0], false, bf[0], (short)0, acc[0], false, false);
        acc[1] = __builtin_amdgcn_wmma_f32_16x16x32_f16(false, af[0], false, bf[1], (short)0, acc[1], false, false);
        acc[2] = __builtin_amdgcn_wmma_f32_16x16x32_f16(false, af[1], false, bf[0], (short)0, acc[2], false, false);
        acc[3] = __builtin_amdgcn_wmma_f32_16x16x32_f16(false, af[1], false, bf[1], (short)0, acc[3], false, false);
    };

    // ---- double-buffered main loop: one barrier per k-step ----
    const int S = (K + BKK - 1) / BKK;
    stage(0, 0);
    for (int s = 1; s < S; ++s) {
        __syncthreads();
        stage(s * BKK, s & 1);        // fill buffer s&1
        mma_step((s - 1) & 1);        // consume previous buffer
    }
    __syncthreads();
    mma_step((S - 1) & 1);

    // ---- epilogue: bias + activation + store ----
#pragma unroll
    for (int mi = 0; mi < 2; ++mi) {
#pragma unroll
        for (int ni = 0; ni < 2; ++ni) {
            const int n = bn0 + nt * 32 + ni * 16 + r;
#pragma unroll
            for (int vr = 0; vr < 8; ++vr) {
                const int m = bm0 + mt * 32 + mi * 16 + vr + 8 * h;
                if (m < M && n < N) {
                    float v = acc[mi * 2 + ni][vr];
                    if (biasMode == 1)      v += bias[m];
                    else if (biasMode == 2) v += bias[n];
                    if (act == 1)      v = (v > 20.f) ? v : log1pf(expf(v));         // softplus
                    else if (act == 2) v = 0.5f * v * (1.f + erff(v * 0.70710678f)); // exact gelu
                    Cp[(long long)m * ldc + n] = v;
                }
            }
        }
    }
}

// a = normalize_rows(softmax_rows(relu(nv1 @ nv2)) + I)   (16x16)
__global__ void adj_kernel(const float* __restrict__ nv1, const float* __restrict__ nv2,
                           float* __restrict__ a_out)
{
    __shared__ float s[NODES][NODES];
    __shared__ float adj[NODES][NODES];
    const int t = threadIdx.x;
    const int v = t >> 4, w = t & 15;
    float acc = 0.f;
    for (int d = 0; d < NODE_DIM; ++d) acc += nv1[v * NODE_DIM + d] * nv2[d * NODES + w];
    s[v][w] = fmaxf(acc, 0.f);
    __syncthreads();
    float mx = -1e30f;
    for (int j = 0; j < NODES; ++j) mx = fmaxf(mx, s[v][j]);
    float den = 0.f;
    for (int j = 0; j < NODES; ++j) den += expf(s[v][j] - mx);
    adj[v][w] = expf(s[v][w] - mx) / den + ((v == w) ? 1.f : 0.f);
    __syncthreads();
    float rs = 0.f;
    for (int j = 0; j < NODES; ++j) rs += adj[v][j];
    a_out[v * NODES + w] = adj[v][w] / rs;
}

// Hcat layout: [v(16)][hop(3)][c(32)][n(2048)]
__global__ void mixprop_kernel(const float* __restrict__ a, float* __restrict__ Hcat,
                               int srcSlot)
{
    const int idx = blockIdx.x * blockDim.x + threadIdx.x;   // c*NTOK + n
    if (idx >= CONV_CH * NTOK) return;
    const long long vs = 3LL * CONV_CH * NTOK;
    const long long ss = (long long)CONV_CH * NTOK;
    float h0v[NODES], hp[NODES];
    for (int w = 0; w < NODES; ++w) {
        h0v[w] = Hcat[w * vs + idx];
        hp[w]  = Hcat[w * vs + srcSlot * ss + idx];
    }
    for (int v = 0; v < NODES; ++v) {
        float accv = 0.f;
        for (int w = 0; w < NODES; ++w) accv += a[v * NODES + w] * hp[w];
        Hcat[v * vs + (srcSlot + 1) * ss + idx] = ALPHA_MP * h0v[v] + (1.f - ALPHA_MP) * accv;
    }
}

// E[(b*64+o)*16 + i] = end_b[o] + sum_{c,l} mixed[i,c,b*64+l] * end_w[o,c,l]
__global__ void end_conv_kernel(const float* __restrict__ mixed,
                                const float* __restrict__ end_w,
                                const float* __restrict__ end_b,
                                float* __restrict__ E)
{
    __shared__ float buf[SKIP_CH * LSEQ];  // 2048 floats
    const int bi = blockIdx.x;             // b*NODES + i
    const int b = bi / NODES, i = bi % NODES;
    const int t = threadIdx.x;             // 64 threads, t == output row o
    for (int j = t; j < SKIP_CH * LSEQ; j += 64) {
        const int c = j >> 6, l = j & 63;
        buf[j] = mixed[(long long)i * (SKIP_CH * NTOK) + (long long)c * NTOK + b * LSEQ + l];
    }
    __syncthreads();
    float acc = end_b[t];
    const float* wrow = end_w + (long long)t * (SKIP_CH * LSEQ);
    for (int j = 0; j < SKIP_CH * LSEQ; ++j) acc += buf[j] * wrow[j];
    E[(long long)(b * LSEQ + t) * NODES + i] = acc;
}

// xg = LayerNorm(x + xlin) * g + b  (one block per token, 256 thr, 2 elems each)
__global__ void residual_ln_kernel(const float* __restrict__ x, const float* __restrict__ xlin,
                                   const float* __restrict__ g, const float* __restrict__ bta,
                                   float* __restrict__ out)
{
    __shared__ float red[256];
    const long long tok = blockIdx.x;
    const int t = threadIdx.x;
    const float v0 = x[tok * DMODEL + t]       + xlin[tok * DMODEL + t];
    const float v1 = x[tok * DMODEL + 256 + t] + xlin[tok * DMODEL + 256 + t];
    red[t] = v0 + v1;
    __syncthreads();
    for (int s = 128; s > 0; s >>= 1) { if (t < s) red[t] += red[t + s]; __syncthreads(); }
    const float mu = red[0] / DMODEL;
    __syncthreads();
    const float d0 = v0 - mu, d1 = v1 - mu;
    red[t] = d0 * d0 + d1 * d1;
    __syncthreads();
    for (int s = 128; s > 0; s >>= 1) { if (t < s) red[t] += red[t + s]; __syncthreads(); }
    const float inv = rsqrtf(red[0] / DMODEL + 1e-5f);
    out[tok * DMODEL + t]       = d0 * inv * g[t]       + bta[t];
    out[tok * DMODEL + 256 + t] = d1 * inv * g[256 + t] + bta[256 + t];
}

// causal depthwise conv1d (width 4) + bias + SiLU, reading xin = XZ[:, :D_INNER]
__global__ void conv1d_silu_kernel(const float* __restrict__ XZ, const float* __restrict__ cw,
                                   const float* __restrict__ cb, float* __restrict__ xc)
{
    const int idx = blockIdx.x * blockDim.x + threadIdx.x;   // b*D_INNER + d
    if (idx >= BATCH * D_INNER) return;
    const int b = idx / D_INNER, d = idx % D_INNER;
    const float w0 = cw[d * 4 + 0], w1 = cw[d * 4 + 1], w2 = cw[d * 4 + 2], w3 = cw[d * 4 + 3];
    const float bias = cb[d];
    float xm3 = 0.f, xm2 = 0.f, xm1 = 0.f;
    for (int l = 0; l < LSEQ; ++l) {
        const float x0 = XZ[(long long)(b * LSEQ + l) * (2 * D_INNER) + d];
        float s = xm3 * w0 + xm2 * w1 + xm1 * w2 + x0 * w3 + bias;
        s = s / (1.f + expf(-s));
        xc[(long long)(b * LSEQ + l) * D_INNER + d] = s;
        xm3 = xm2; xm2 = xm1; xm1 = x0;
    }
}

// selective scan: one thread per (b, d); 16-state recurrence in registers
__global__ void scan_kernel(const float* __restrict__ DT, const float* __restrict__ XDBL,
                            const float* __restrict__ xc, const float* __restrict__ XZ,
                            const float* __restrict__ A_log, const float* __restrict__ Dp,
                            float* __restrict__ Y)
{
    const int idx = blockIdx.x * blockDim.x + threadIdx.x;   // b*D_INNER + d
    if (idx >= BATCH * D_INNER) return;
    const int b = idx / D_INNER, d = idx % D_INNER;
    float negA[D_STATE], hs[D_STATE];
#pragma unroll
    for (int s = 0; s < D_STATE; ++s) { negA[s] = -expf(A_log[d * D_STATE + s]); hs[s] = 0.f; }
    const float Dd = Dp[d];
    for (int l = 0; l < LSEQ; ++l) {
        const long long tok = b * LSEQ + l;
        const float dt = DT[tok * D_INNER + d];
        const float u  = xc[tok * D_INNER + d];
        const float* xd = XDBL + tok * (DT_RANK + 2 * D_STATE);
        float y = 0.f;
#pragma unroll
        for (int s = 0; s < D_STATE; ++s) {
            const float dA = expf(dt * negA[s]);
            hs[s] = dA * hs[s] + dt * xd[DT_RANK + s] * u;
            y += hs[s] * xd[DT_RANK + D_STATE + s];
        }
        y += u * Dd;
        const float z = XZ[tok * (2 * D_INNER) + D_INNER + d];
        y *= z / (1.f + expf(-z));
        Y[tok * D_INNER + d] = y;
    }
}

// out = gelu(LayerNorm(O)*g + b) + xg
__global__ void final_kernel(const float* __restrict__ O, const float* __restrict__ g,
                             const float* __restrict__ bta, const float* __restrict__ xg,
                             float* __restrict__ out)
{
    __shared__ float red[256];
    const long long tok = blockIdx.x;
    const int t = threadIdx.x;
    const float v0 = O[tok * DMODEL + t];
    const float v1 = O[tok * DMODEL + 256 + t];
    red[t] = v0 + v1;
    __syncthreads();
    for (int s = 128; s > 0; s >>= 1) { if (t < s) red[t] += red[t + s]; __syncthreads(); }
    const float mu = red[0] / DMODEL;
    __syncthreads();
    const float d0 = v0 - mu, d1 = v1 - mu;
    red[t] = d0 * d0 + d1 * d1;
    __syncthreads();
    for (int s = 128; s > 0; s >>= 1) { if (t < s) red[t] += red[t + s]; __syncthreads(); }
    const float inv = rsqrtf(red[0] / DMODEL + 1e-5f);
    float a0 = d0 * inv * g[t]       + bta[t];
    float a1 = d1 * inv * g[256 + t] + bta[256 + t];
    a0 = 0.5f * a0 * (1.f + erff(a0 * 0.70710678f));
    a1 = 0.5f * a1 * (1.f + erff(a1 * 0.70710678f));
    out[tok * DMODEL + t]       = a0 + xg[tok * DMODEL + t];
    out[tok * DMODEL + 256 + t] = a1 + xg[tok * DMODEL + 256 + t];
}

extern "C" void kernel_launch(void* const* d_in, const int* in_sizes, int n_in,
                              void* d_out, int out_size, void* d_ws, size_t ws_size,
                              hipStream_t stream)
{
    const float* x        = (const float*)d_in[0];
    const float* nv1      = (const float*)d_in[1];
    const float* nv2      = (const float*)d_in[2];
    const float* start_w  = (const float*)d_in[3];
    const float* start_b  = (const float*)d_in[4];
    const float* mix_w    = (const float*)d_in[5];
    const float* mix_b    = (const float*)d_in[6];
    const float* end_w    = (const float*)d_in[7];
    const float* end_b    = (const float*)d_in[8];
    const float* lin_w    = (const float*)d_in[9];
    const float* lin_b    = (const float*)d_in[10];
    const float* gnorm_g  = (const float*)d_in[11];
    const float* gnorm_b  = (const float*)d_in[12];
    const float* in_proj_w= (const float*)d_in[13];
    const float* conv_w   = (const float*)d_in[14];
    const float* conv_b   = (const float*)d_in[15];
    const float* xproj_w  = (const float*)d_in[16];
    const float* dtproj_w = (const float*)d_in[17];
    const float* dtproj_b = (const float*)d_in[18];
    const float* A_log    = (const float*)d_in[19];
    const float* Dp       = (const float*)d_in[20];
    const float* out_proj_w=(const float*)d_in[21];
    const float* norm_g   = (const float*)d_in[22];
    const float* norm_b   = (const float*)d_in[23];

    float* ws   = (float*)d_ws;
    float* xg   = ws;                       // 1,048,576 (persistent)
    float* adja = ws + 1048576;             // 256
    float* U    = ws + 1048832;             // union scratch
    // --- graph phase layout in U ---
    float* Hcat  = U;                       // 16*3*32*2048 = 3,145,728
    float* mixed = U + 3145728;             // 16*32*2048   = 1,048,576
    float* E     = mixed + 1048576;         // 2048*16      = 32,768
    float* xlin  = E + 32768;               // 2048*512     = 1,048,576
    // --- mamba phase layout in U (graph scratch dead after residual_ln) ---
    float* XZ    = U;                       // 2048*2048 = 4,194,304
    float* xcb   = XZ + 4194304;            // 2048*1024 = 2,097,152
    float* XDBL  = xcb + 2097152;           // 2048*64   = 131,072
    float* DTm   = XDBL + 131072;           // 2048*1024 = 2,097,152
    float* Ybuf  = DTm + 2097152;           // 2048*1024 = 2,097,152
    float* Obuf  = Ybuf + 2097152;          // 2048*512  = 1,048,576

    // 1) adaptive adjacency
    adj_kernel<<<1, 256, 0, stream>>>(nv1, nv2, adja);

    // 2) start_conv as 16 shifted GEMMs: C_i[32x2048] = start_w[32x497] * x_shifted^T
    wmma_gemm<<<dim3((NTOK + BN - 1) / BN, 1, NODES), 256, 0, stream>>>(
        start_w, x, Hcat, start_b,
        CONV_CH, NTOK, KH, KH, DMODEL, NTOK,
        /*transB=*/1, /*bias per-m*/1, /*act*/0, /*vecA*/0, /*vecB*/0,
        0LL, /*shift by node idx*/1LL, 3LL * CONV_CH * NTOK);

    // 3) two mixprop hops
    mixprop_kernel<<<(CONV_CH * NTOK + 255) / 256, 256, 0, stream>>>(adja, Hcat, 0);
    mixprop_kernel<<<(CONV_CH * NTOK + 255) / 256, 256, 0, stream>>>(adja, Hcat, 1);

    // 4) mix 1x1 conv + gelu: per node v, C_v[32x2048] = mix_w[32x96] * Hcat_v[96x2048]
    wmma_gemm<<<dim3((NTOK + BN - 1) / BN, 1, NODES), 256, 0, stream>>>(
        mix_w, Hcat, mixed, mix_b,
        SKIP_CH, NTOK, 3 * CONV_CH, 3 * CONV_CH, NTOK, NTOK,
        /*transB=*/0, /*bias per-m*/1, /*act gelu*/2, /*vecA*/1, /*vecB*/1,
        0LL, 3LL * CONV_CH * NTOK, (long long)SKIP_CH * NTOK);

    // 5) end_conv
    end_conv_kernel<<<BATCH * NODES, 64, 0, stream>>>(mixed, end_w, end_b, E);

    // 6) lin: xlin[2048x512] = E[2048x16] * lin_w^T + lin_b
    wmma_gemm<<<dim3(DMODEL / BN, NTOK / BM, 1), 256, 0, stream>>>(
        E, lin_w, xlin, lin_b,
        NTOK, DMODEL, NODES, NODES, NODES, DMODEL,
        1, /*bias per-n*/2, 0, 1, 1, 0LL, 0LL, 0LL);

    // 7) xg = LN(x + xlin)
    residual_ln_kernel<<<NTOK, 256, 0, stream>>>(x, xlin, gnorm_g, gnorm_b, xg);

    // 8) in_proj: XZ[2048x2048] = xg * in_proj_w^T
    wmma_gemm<<<dim3(2 * D_INNER / BN, NTOK / BM, 1), 256, 0, stream>>>(
        xg, in_proj_w, XZ, nullptr,
        NTOK, 2 * D_INNER, DMODEL, DMODEL, DMODEL, 2 * D_INNER,
        1, 0, 0, 1, 1, 0LL, 0LL, 0LL);

    // 9) depthwise causal conv + SiLU
    conv1d_silu_kernel<<<(BATCH * D_INNER + 255) / 256, 256, 0, stream>>>(XZ, conv_w, conv_b, xcb);

    // 10) x_proj: XDBL[2048x64] = xc * xproj_w^T  (N=64 < BN, clamped)
    wmma_gemm<<<dim3(1, NTOK / BM, 1), 256, 0, stream>>>(
        xcb, xproj_w, XDBL, nullptr,
        NTOK, DT_RANK + 2 * D_STATE, D_INNER, D_INNER, D_INNER, DT_RANK + 2 * D_STATE,
        1, 0, 0, 1, 1, 0LL, 0LL, 0LL);

    // 11) dt_proj + softplus: DT[2048x1024] = XDBL[:, :32] * dtproj_w^T + dtproj_b
    wmma_gemm<<<dim3(D_INNER / BN, NTOK / BM, 1), 256, 0, stream>>>(
        XDBL, dtproj_w, DTm, dtproj_b,
        NTOK, D_INNER, DT_RANK, DT_RANK + 2 * D_STATE, DT_RANK, D_INNER,
        1, /*bias per-n*/2, /*softplus*/1, 1, 1, 0LL, 0LL, 0LL);

    // 12) selective scan + D skip + z gate
    scan_kernel<<<(BATCH * D_INNER + 255) / 256, 256, 0, stream>>>(
        DTm, XDBL, xcb, XZ, A_log, Dp, Ybuf);

    // 13) out_proj: O[2048x512] = Y * out_proj_w^T
    wmma_gemm<<<dim3(DMODEL / BN, NTOK / BM, 1), 256, 0, stream>>>(
        Ybuf, out_proj_w, Obuf, nullptr,
        NTOK, DMODEL, D_INNER, D_INNER, D_INNER, DMODEL,
        1, 0, 0, 1, 1, 0LL, 0LL, 0LL);

    // 14) final LN + gelu + residual
    final_kernel<<<NTOK, 256, 0, stream>>>(Obuf, norm_g, norm_b, xg, (float*)d_out);
}
